// DIEN_66314295050880
// MI455X (gfx1250) — matrix operand order repaired and checked by reading, not compiled, not run on hardware
//
#include <hip/hip_runtime.h>
#include <hip/hip_bf16.h>

typedef __attribute__((ext_vector_type(16))) _Float16 v16h;
typedef __attribute__((ext_vector_type(8)))  float    v8f;

#define WMMA_F16(a, b, c) \
  __builtin_amdgcn_wmma_f32_16x16x32_f16(false, (a), false, (b), (short)0, (c), false, false)

#define BDIM 256   // batch
#define TDIM 256   // seq len
#define BT   65536 // B*T
#define MD   256   // GRU hidden

__device__ __forceinline__ float sigf(float x) { return 1.f / (1.f + __expf(-x)); }

// ---- WMMA fragment helpers (contiguous-k layouts, wave32) -------------------
// A 16x32 f16: lane&15 = row M; lane>>4 selects K-halves {0..15} vs {16..31}
__device__ __forceinline__ int a_kidx(int lane, int j) {
  return (((lane >> 4) & 1) * 16) + j;
}
// plain row-major A tile
__device__ __forceinline__ v16h load_a(const _Float16* X, int ld, int r0, int k0, int lane) {
  const _Float16* p = X + (r0 + (lane & 15)) * ld + k0;
  v16h a;
#pragma unroll
  for (int j = 0; j < 16; ++j) a[j] = p[a_kidx(lane, j)];
  return a;
}
// A tile synthesizing DIN features [q, k, q-k, q*k] over 4 segments of 256
__device__ __forceinline__ v16h load_a_din(const _Float16* inter, const _Float16* tgt,
                                           int r0, int k0, int lane) {
  int r = r0 + (lane & 15);
  int b = r >> 8;
  v16h a;
#pragma unroll
  for (int j = 0; j < 16; ++j) {
    int kk = k0 + a_kidx(lane, j);
    int seg = kk >> 8, c = kk & 255;
    _Float16 q = tgt[(b << 8) + c];
    _Float16 kv = inter[(r << 8) + c];
    a[j] = (seg == 0) ? q : (seg == 1) ? kv
         : (seg == 2) ? (_Float16)(q - kv) : (_Float16)(q * kv);
  }
  return a;
}
// A tile synthesizing pooling features [q, k, q-k, q*k] over 4 segments of 128
__device__ __forceinline__ v16h load_a_ci(const _Float16* seq, const _Float16* tgt, int off,
                                          int r0, int k0, int lane) {
  int r = r0 + (lane & 15);
  int b = r >> 8;
  v16h a;
#pragma unroll
  for (int j = 0; j < 16; ++j) {
    int kk = k0 + a_kidx(lane, j);
    int seg = kk >> 7, c = (kk & 127) + off;
    _Float16 q = tgt[(b << 8) + c];
    _Float16 bv = seq[(r << 8) + c];
    a[j] = (seg == 0) ? q : (seg == 1) ? bv
         : (seg == 2) ? (_Float16)(q - bv) : (_Float16)(q * bv);
  }
  return a;
}
// B 32x16 f16 = W^T tile; W row-major [N, K]; lane&15 = col N, 16 contiguous K halves
__device__ __forceinline__ v16h load_b(const _Float16* W, int ld, int n0, int k0, int lane) {
  const _Float16* p = W + (n0 + (lane & 15)) * ld + k0 + ((lane >> 4) & 1) * 16;
  v16h b;
#pragma unroll
  for (int j = 0; j < 16; ++j) b[j] = p[j];
  return b;
}

// ---- CDNA5 async global->LDS copy (ASYNCcnt-tracked); one 512B row per wave
__device__ __forceinline__ void async_row_to_lds(const _Float16* src, unsigned lds_off,
                                                 int lane) {
  unsigned la = lds_off + (unsigned)(lane * 16);
  unsigned long long ga =
      (unsigned long long)(uintptr_t)src + (unsigned long long)(lane * 16);
  asm volatile("global_load_async_to_lds_b128 %0, %1, off" :: "v"(la), "v"(ga) : "memory");
}
__device__ __forceinline__ void wait_asynccnt_le2() {
  asm volatile("s_wait_asynccnt 0x2" ::: "memory");
}

// ---- small utility kernels -------------------------------------------------
__global__ void k_cvt(const float* __restrict__ s, _Float16* __restrict__ d, int n) {
  int i = blockIdx.x * 256 + threadIdx.x;
  if (i < n) d[i] = (_Float16)s[i];
}

__global__ void k_embed_seq(const int* __restrict__ iseq, const int* __restrict__ cseq,
                            const float* __restrict__ itab, const float* __restrict__ ctab,
                            _Float16* __restrict__ seq) {
  int idx = blockIdx.x * 256 + threadIdx.x;   // BT*256
  int r = idx >> 8, c = idx & 255;
  float v = (c < 128) ? itab[iseq[r] * 128 + c] : ctab[cseq[r] * 128 + (c - 128)];
  seq[idx] = (_Float16)v;
}

__global__ void k_embed_tgt(const int* __restrict__ ti, const int* __restrict__ tc,
                            const float* __restrict__ itab, const float* __restrict__ ctab,
                            _Float16* __restrict__ t16, float* __restrict__ t32) {
  int idx = blockIdx.x * 256 + threadIdx.x;   // B*256
  int b = idx >> 8, c = idx & 255;
  float v = (c < 128) ? itab[ti[b] * 128 + c] : ctab[tc[b] * 128 + (c - 128)];
  t16[idx] = (_Float16)v;
  t32[idx] = v;
}

// ---- masked GRU scan: 16 batch rows per WG, WMMA [16,256]x[256,768] per step
// x-operand tiles double-buffered in LDS via async global->LDS prefetch.
__global__ void __launch_bounds__(256) k_gru(
    const _Float16* __restrict__ seq, const _Float16* __restrict__ Wih,
    const _Float16* __restrict__ Whh, const float* __restrict__ bih,
    const float* __restrict__ bhh, const int* __restrict__ lengths,
    _Float16* __restrict__ interest) {
  __shared__ _Float16 hf16[16 * 256];
  __shared__ _Float16 sbuf[16 * 768];      // r,z: gx+gh ; n: gx part
  __shared__ _Float16 snh[16 * 256];       // n: gh part
  __shared__ _Float16 xbuf[2 * 16 * 256];  // double-buffered x tile
  int b0 = blockIdx.x * 16;
  int tid = threadIdx.x, lane = tid & 31;
  int wave = __builtin_amdgcn_readfirstlane(tid >> 5);   // wave-uniform -> SGPR
  int row = tid >> 4, cb = tid & 15;
  unsigned xbase = (unsigned)(uintptr_t)xbuf;
  float h[16];
#pragma unroll
  for (int i = 0; i < 16; ++i) { h[i] = 0.f; hf16[row * 256 + cb + 16 * i] = (_Float16)0.f; }
  int len = lengths[b0 + row];
  // prologue: async-prefetch tile t=0 into buffer 0 (rows 2w, 2w+1 per wave)
#pragma unroll
  for (int rr = 0; rr < 2; ++rr) {
    int r = wave * 2 + rr;
    async_row_to_lds(seq + ((((b0 + r) << 8) + 0) << 8), xbase + (unsigned)(r * 512), lane);
  }
  for (int t = 0; t < TDIM; ++t) {
    int sel = t & 1;
    int tp = (t < TDIM - 1) ? t + 1 : TDIM - 1;
#pragma unroll
    for (int rr = 0; rr < 2; ++rr) {   // prefetch next step's x tile
      int r = wave * 2 + rr;
      async_row_to_lds(seq + ((((b0 + r) << 8) + tp) << 8),
                       xbase + (unsigned)(((sel ^ 1) * 16 + r) * 512), lane);
    }
    wait_asynccnt_le2();               // tile t resident in LDS
    __syncthreads();
    const _Float16* xt = xbuf + sel * 4096;
    for (int ct = wave; ct < 48; ct += 8) {
      int n0 = ct * 16;
      int n = lane & 15, mb = (lane >> 4) * 8;
      if (n0 < 512) {            // r,z gates: x and h parts sum
        v8f acc = {};
        for (int kt = 0; kt < 8; ++kt) {
          acc = WMMA_F16(load_a(hf16, 256, 0, kt * 32, lane),
                         load_b(Whh, 256, n0, kt * 32, lane), acc);
          acc = WMMA_F16(load_a(xt, 256, 0, kt * 32, lane),
                         load_b(Wih, 256, n0, kt * 32, lane), acc);
        }
        float bs = bih[n0 + n] + bhh[n0 + n];
#pragma unroll
        for (int v = 0; v < 8; ++v)
          sbuf[(mb + v) * 768 + n0 + n] = (_Float16)(acc[v] + bs);
      } else {                   // n gate: keep x/h parts separate
        v8f ax = {}, ah = {};
        for (int kt = 0; kt < 8; ++kt) {
          ah = WMMA_F16(load_a(hf16, 256, 0, kt * 32, lane),
                        load_b(Whh, 256, n0, kt * 32, lane), ah);
          ax = WMMA_F16(load_a(xt, 256, 0, kt * 32, lane),
                        load_b(Wih, 256, n0, kt * 32, lane), ax);
        }
        float bx = bih[n0 + n], bh = bhh[n0 + n];
#pragma unroll
        for (int v = 0; v < 8; ++v) {
          sbuf[(mb + v) * 768 + n0 + n] = (_Float16)(ax[v] + bx);
          snh[(mb + v) * 256 + (n0 - 512) + n] = (_Float16)(ah[v] + bh);
        }
      }
    }
    __syncthreads();
    bool mt = t < len;
#pragma unroll
    for (int i = 0; i < 16; ++i) {
      int c = cb + 16 * i;
      float r = sigf((float)sbuf[row * 768 + c]);
      float z = sigf((float)sbuf[row * 768 + 256 + c]);
      float n = tanhf((float)sbuf[row * 768 + 512 + c] + r * (float)snh[row * 256 + c]);
      float hy = (1.f - z) * n + z * h[i];
      if (mt) h[i] = hy;
      interest[((((b0 + row) << 8) + t) << 8) + c] = (_Float16)(mt ? hy : 0.f);
      hf16[row * 256 + c] = (_Float16)h[i];
    }
  }
}

// ---- AGRU scan (chunks u(skip), r, n), attentional update ------------------
__global__ void __launch_bounds__(256) k_agru(
    const _Float16* __restrict__ inter, const _Float16* __restrict__ Wx,
    const _Float16* __restrict__ Wh, const float* __restrict__ bx,
    const float* __restrict__ bh, const float* __restrict__ attn,
    const int* __restrict__ lengths, float* __restrict__ hT) {
  __shared__ _Float16 hf16[16 * 256];
  __shared__ _Float16 sbuf[16 * 512];      // local col: 0..255 = r pre-act, 256..511 = xn
  __shared__ _Float16 snh[16 * 256];       // hn
  __shared__ _Float16 xbuf[2 * 16 * 256];  // double-buffered interest tile
  int b0 = blockIdx.x * 16;
  int tid = threadIdx.x, lane = tid & 31;
  int wave = __builtin_amdgcn_readfirstlane(tid >> 5);
  int row = tid >> 4, cb = tid & 15;
  unsigned xbase = (unsigned)(uintptr_t)xbuf;
  float h[16];
#pragma unroll
  for (int i = 0; i < 16; ++i) { h[i] = 0.f; hf16[row * 256 + cb + 16 * i] = (_Float16)0.f; }
  int len = lengths[b0 + row];
#pragma unroll
  for (int rr = 0; rr < 2; ++rr) {
    int r = wave * 2 + rr;
    async_row_to_lds(inter + ((((b0 + r) << 8) + 0) << 8), xbase + (unsigned)(r * 512), lane);
  }
  for (int t = 0; t < TDIM; ++t) {
    int sel = t & 1;
    int tp = (t < TDIM - 1) ? t + 1 : TDIM - 1;
#pragma unroll
    for (int rr = 0; rr < 2; ++rr) {
      int r = wave * 2 + rr;
      async_row_to_lds(inter + ((((b0 + r) << 8) + tp) << 8),
                       xbase + (unsigned)(((sel ^ 1) * 16 + r) * 512), lane);
    }
    wait_asynccnt_le2();
    __syncthreads();
    const _Float16* xt = xbuf + sel * 4096;
    for (int ct = 16 + wave; ct < 48; ct += 8) {   // skip 'u' chunk (cols 0..255)
      int n0 = ct * 16;
      int n = lane & 15, mb = (lane >> 4) * 8;
      if (n0 < 512) {            // r gate
        v8f acc = {};
        for (int kt = 0; kt < 8; ++kt) {
          acc = WMMA_F16(load_a(hf16, 256, 0, kt * 32, lane),
                         load_b(Wh, 256, n0, kt * 32, lane), acc);
          acc = WMMA_F16(load_a(xt, 256, 0, kt * 32, lane),
                         load_b(Wx, 256, n0, kt * 32, lane), acc);
        }
        float bs = bx[n0 + n] + bh[n0 + n];
#pragma unroll
        for (int v = 0; v < 8; ++v)
          sbuf[(mb + v) * 512 + (n0 - 256) + n] = (_Float16)(acc[v] + bs);
      } else {                   // n gate
        v8f ax = {}, ah = {};
        for (int kt = 0; kt < 8; ++kt) {
          ah = WMMA_F16(load_a(hf16, 256, 0, kt * 32, lane),
                        load_b(Wh, 256, n0, kt * 32, lane), ah);
          ax = WMMA_F16(load_a(xt, 256, 0, kt * 32, lane),
                        load_b(Wx, 256, n0, kt * 32, lane), ax);
        }
        float bxv = bx[n0 + n], bhv = bh[n0 + n];
#pragma unroll
        for (int v = 0; v < 8; ++v) {
          sbuf[(mb + v) * 512 + (n0 - 256) + n] = (_Float16)(ax[v] + bxv);
          snh[(mb + v) * 256 + (n0 - 512) + n] = (_Float16)(ah[v] + bhv);
        }
      }
    }
    __syncthreads();
    bool mt = t < len;
    float at = attn[((b0 + row) << 8) + t];
#pragma unroll
    for (int i = 0; i < 16; ++i) {
      int c = cb + 16 * i;
      float r = sigf((float)sbuf[row * 512 + c]);
      float n = tanhf((float)sbuf[row * 512 + 256 + c] + r * (float)snh[row * 256 + c]);
      float hy = h[i] + at * (n - h[i]);
      if (mt) h[i] = hy;
      hf16[row * 256 + c] = (_Float16)h[i];
    }
  }
  __syncthreads();
#pragma unroll
  for (int i = 0; i < 16; ++i) hT[((b0 + row) << 8) + cb + 16 * i] = h[i];
}

// ---- WMMA GEMMs over BT rows ----------------------------------------------
// DIN attention layer1: fused 8d features, K=1024 -> N=512
__global__ void __launch_bounds__(256) k_gemm_din(
    const _Float16* __restrict__ inter, const _Float16* __restrict__ tgt,
    const _Float16* __restrict__ W, const float* __restrict__ bias,
    _Float16* __restrict__ Z) {
  int r0 = blockIdx.x * 16;
  int lane = threadIdx.x & 31;
  int wave = __builtin_amdgcn_readfirstlane(threadIdx.x >> 5);
  for (int ct = wave; ct < 32; ct += 8) {
    int n0 = ct * 16;
    v8f acc = {};
    for (int kt = 0; kt < 32; ++kt)
      acc = WMMA_F16(load_a_din(inter, tgt, r0, kt * 32, lane),
                     load_b(W, 1024, n0, kt * 32, lane), acc);
    int n = lane & 15, mb = (lane >> 4) * 8;
    float bv = bias[n0 + n];
#pragma unroll
    for (int v = 0; v < 8; ++v)
      Z[(r0 + mb + v) * 512 + n0 + n] = (_Float16)(acc[v] + bv);
  }
}

// plain GEMM: Z[BT,N] = X[BT,K] @ W^T + b   (K%32==0, N%16==0)
__global__ void __launch_bounds__(256) k_gemm_plain(
    const _Float16* __restrict__ X, int K, const _Float16* __restrict__ W, int N,
    const float* __restrict__ bias, _Float16* __restrict__ Z) {
  int r0 = blockIdx.x * 16;
  int lane = threadIdx.x & 31;
  int wave = __builtin_amdgcn_readfirstlane(threadIdx.x >> 5);
  int NT = N >> 4, KT = K >> 5;
  for (int ct = wave; ct < NT; ct += 8) {
    int n0 = ct * 16;
    v8f acc = {};
    for (int kt = 0; kt < KT; ++kt)
      acc = WMMA_F16(load_a(X, K, r0, kt * 32, lane),
                     load_b(W, K, n0, kt * 32, lane), acc);
    int n = lane & 15, mb = (lane >> 4) * 8;
    float bv = bias[n0 + n];
#pragma unroll
    for (int v = 0; v < 8; ++v)
      Z[(r0 + mb + v) * N + n0 + n] = (_Float16)(acc[v] + bv);
  }
}

// pooling layer1: fused 4d features (K=512) -> N=64; 32 rows per block
__global__ void __launch_bounds__(256) k_gemm_ci(
    const _Float16* __restrict__ seq, const _Float16* __restrict__ tgt, int off,
    const _Float16* __restrict__ W, const float* __restrict__ bias,
    _Float16* __restrict__ Z) {
  int lane = threadIdx.x & 31;
  int wave = __builtin_amdgcn_readfirstlane(threadIdx.x >> 5);
  int r0 = blockIdx.x * 32 + (wave >> 2) * 16;
  int n0 = (wave & 3) * 16;
  v8f acc = {};
  for (int kt = 0; kt < 16; ++kt)
    acc = WMMA_F16(load_a_ci(seq, tgt, off, r0, kt * 32, lane),
                   load_b(W, 512, n0, kt * 32, lane), acc);
  int n = lane & 15, mb = (lane >> 4) * 8;
  float bv = bias[n0 + n];
#pragma unroll
  for (int v = 0; v < 8; ++v)
    Z[(r0 + mb + v) * 64 + n0 + n] = (_Float16)(acc[v] + bv);
}

// ---- Dice: deterministic per-channel stats + in-place activation -----------
__global__ void k_colstats(const _Float16* __restrict__ Z, int C, float* __restrict__ st) {
  __shared__ float s1[256], s2[256];
  int c = blockIdx.x, tid = threadIdx.x;
  float a = 0.f, b = 0.f;
  for (int r = tid; r < BT; r += 256) {
    float v = (float)Z[r * C + c];
    a += v; b += v * v;
  }
  s1[tid] = a; s2[tid] = b;
  __syncthreads();
  for (int s = 128; s; s >>= 1) {
    if (tid < s) { s1[tid] += s1[tid + s]; s2[tid] += s2[tid + s]; }
    __syncthreads();
  }
  if (!tid) { st[c] = s1[0]; st[C + c] = s2[0]; }
}

__global__ void k_dice(_Float16* __restrict__ Z, int C, const float* __restrict__ st,
                       const float* __restrict__ alpha) {
  int idx = blockIdx.x * 256 + threadIdx.x;
  int c = idx % C;
  float mu = st[c] * (1.f / (float)BT);
  float var = st[C + c] * (1.f / (float)BT) - mu * mu;
  float x = (float)Z[idx];
  float p = sigf((x - mu) * rsqrtf(var + 1e-9f));
  Z[idx] = (_Float16)((p + alpha[c] * (1.f - p)) * x);
}

// final 1-wide linear: out[r] = dot(X[r,:C], w) + b   (one wave per row)
__global__ void k_dot(const _Float16* __restrict__ X, int C, const float* __restrict__ w,
                      const float* __restrict__ b, float* __restrict__ out) {
  int row = blockIdx.x * 8 + (threadIdx.x >> 5);
  int lane = threadIdx.x & 31;
  float acc = 0.f;
  for (int c = lane; c < C; c += 32) acc += (float)X[row * C + c] * w[c];
  for (int off = 16; off; off >>= 1) acc += __shfl_down(acc, off, 32);
  if (!lane) out[row] = acc + b[0];
}

// ---- softmaxes -------------------------------------------------------------
__global__ void k_softmax_att(const float* __restrict__ score, const int* __restrict__ len,
                              float* __restrict__ attn) {
  __shared__ float red[256];
  int b = blockIdx.x, t = threadIdx.x;
  float s = (t < len[b]) ? score[(b << 8) + t] : -1e9f;
  red[t] = s; __syncthreads();
  for (int st = 128; st; st >>= 1) { if (t < st) red[t] = fmaxf(red[t], red[t + st]); __syncthreads(); }
  float mx = red[0]; __syncthreads();
  float e = __expf(s - mx);
  red[t] = e; __syncthreads();
  for (int st = 128; st; st >>= 1) { if (t < st) red[t] += red[t + st]; __syncthreads(); }
  attn[(b << 8) + t] = e / red[0];
}

__global__ void k_pool(const float* __restrict__ score, const int* __restrict__ len,
                       const _Float16* __restrict__ seq, int off, float* __restrict__ pooled) {
  __shared__ float w[256], red[256];
  int b = blockIdx.x, t = threadIdx.x;
  float s = score[(b << 8) + t] - ((t < len[b]) ? 0.f : 1e4f);
  red[t] = s; __syncthreads();
  for (int st = 128; st; st >>= 1) { if (t < st) red[t] = fmaxf(red[t], red[t + st]); __syncthreads(); }
  float mx = red[0]; __syncthreads();
  float e = __expf(s - mx);
  red[t] = e; __syncthreads();
  for (int st = 128; st; st >>= 1) { if (t < st) red[t] += red[t + st]; __syncthreads(); }
  w[t] = e / red[0]; __syncthreads();
  if (t < 128) {
    float acc = 0.f;
    for (int tt = 0; tt < 256; ++tt)
      acc += (float)seq[(((b << 8) + tt) << 8) + off + t] * w[tt];
    pooled[b * 128 + t] = acc;
  }
}

// ---- final feature + fp32 MLP ---------------------------------------------
__global__ void k_feat(const float* __restrict__ tgt, const float* __restrict__ ip,
                       const float* __restrict__ cp, const float* __restrict__ hT,
                       float* __restrict__ feat) {
  int idx = blockIdx.x * 256 + threadIdx.x;  // B*1024
  int b = idx >> 10, j = idx & 1023;
  float v;
  if (j < 256)      v = tgt[(b << 8) + j];
  else if (j < 384) v = ip[b * 128 + (j - 256)];
  else if (j < 512) v = cp[b * 128 + (j - 384)];
  else if (j < 768) v = hT[(b << 8) + (j - 512)];
  else              v = tgt[(b << 8) + (j - 768)] * hT[(b << 8) + (j - 768)];
  feat[idx] = v;
}

__global__ void __launch_bounds__(256) k_mlp(
    const float* __restrict__ feat, const float* __restrict__ W1, const float* __restrict__ b1,
    const float* __restrict__ W2, const float* __restrict__ b2,
    const float* __restrict__ W3, const float* __restrict__ b3, float* __restrict__ out) {
  __shared__ float h1[256], h2[128], red[256];
  int b = blockIdx.x, tid = threadIdx.x;
  const float* f = feat + b * 1024;
  const float* w = W1 + tid * 1024;
  float a = 0.f;
  for (int k = 0; k < 1024; ++k) a += f[k] * w[k];
  h1[tid] = fmaxf(a + b1[tid], 0.f);
  __syncthreads();
  if (tid < 128) {
    const float* w2 = W2 + tid * 256;
    float a2 = 0.f;
    for (int k = 0; k < 256; ++k) a2 += h1[k] * w2[k];
    h2[tid] = fmaxf(a2 + b2[tid], 0.f);
  }
  __syncthreads();
  red[tid] = (tid < 128) ? h2[tid] * W3[tid] : 0.f;
  __syncthreads();
  for (int s = 128; s; s >>= 1) { if (tid < s) red[tid] += red[tid + s]; __syncthreads(); }
  if (!tid) out[b] = red[0] + b3[0];
}

// ---------------------------------------------------------------------------
extern "C" void kernel_launch(void* const* d_in, const int* in_sizes, int n_in,
                              void* d_out, int out_size, void* d_ws, size_t ws_size,
                              hipStream_t stream) {
  // inputs (setup_inputs dict order; params flattened in insertion order)
  const int* item_seq = (const int*)d_in[0];
  const int* cate_seq = (const int*)d_in[1];
  const int* tgt_item = (const int*)d_in[2];
  const int* tgt_cate = (const int*)d_in[3];
  const int* lengths  = (const int*)d_in[4];
  const float* item_table = (const float*)d_in[5];
  const float* cate_table = (const float*)d_in[6];
  const float* gru_Wih = (const float*)d_in[7];
  const float* gru_Whh = (const float*)d_in[8];
  const float* gru_bih = (const float*)d_in[9];
  const float* gru_bhh = (const float*)d_in[10];
  const float* agru_Wx = (const float*)d_in[11];
  const float* agru_bx = (const float*)d_in[12];
  const float* agru_Wh = (const float*)d_in[13];
  const float* agru_bh = (const float*)d_in[14];
  const float* att_W1 = (const float*)d_in[15];
  const float* att_b1 = (const float*)d_in[16];
  const float* att_a1 = (const float*)d_in[17];
  const float* att_W2 = (const float*)d_in[18];
  const float* att_b2 = (const float*)d_in[19];
  const float* att_a2 = (const float*)d_in[20];
  const float* att_W3 = (const float*)d_in[21];
  const float* att_b3 = (const float*)d_in[22];
  const float* pi_W1 = (const float*)d_in[23];
  const float* pi_b1 = (const float*)d_in[24];
  const float* pi_a1 = (const float*)d_in[25];
  const float* pi_W2 = (const float*)d_in[26];
  const float* pi_b2 = (const float*)d_in[27];
  const float* pi_a2 = (const float*)d_in[28];
  const float* pi_W3 = (const float*)d_in[29];
  const float* pi_b3 = (const float*)d_in[30];
  const float* pc_W1 = (const float*)d_in[31];
  const float* pc_b1 = (const float*)d_in[32];
  const float* pc_a1 = (const float*)d_in[33];
  const float* pc_W2 = (const float*)d_in[34];
  const float* pc_b2 = (const float*)d_in[35];
  const float* pc_a2 = (const float*)d_in[36];
  const float* pc_W3 = (const float*)d_in[37];
  const float* pc_b3 = (const float*)d_in[38];
  const float* mlp_W1 = (const float*)d_in[39];
  const float* mlp_b1 = (const float*)d_in[40];
  const float* mlp_W2 = (const float*)d_in[41];
  const float* mlp_b2 = (const float*)d_in[42];
  const float* mlp_W3 = (const float*)d_in[43];
  const float* mlp_b3 = (const float*)d_in[44];

  // workspace layout
  char* base = (char*)d_ws;
  size_t off = 0;
  auto alloc = [&](size_t bytes) -> char* {
    char* p = base + off;
    off += (bytes + 255) & ~(size_t)255;
    return p;
  };
  _Float16* seq16   = (_Float16*)alloc((size_t)BT * 256 * 2);
  _Float16* inter16 = (_Float16*)alloc((size_t)BT * 256 * 2);
  _Float16* tgt16   = (_Float16*)alloc((size_t)BDIM * 256 * 2);
  float*    tgt32   = (float*)   alloc((size_t)BDIM * 256 * 4);
  _Float16* wih16   = (_Float16*)alloc(768 * 256 * 2);
  _Float16* whh16   = (_Float16*)alloc(768 * 256 * 2);
  _Float16* wax16   = (_Float16*)alloc(768 * 256 * 2);
  _Float16* wah16   = (_Float16*)alloc(768 * 256 * 2);
  _Float16* wa116   = (_Float16*)alloc(512 * 1024 * 2);
  _Float16* wa216   = (_Float16*)alloc(256 * 512 * 2);
  _Float16* wpi116  = (_Float16*)alloc(64 * 512 * 2);
  _Float16* wpi216  = (_Float16*)alloc(32 * 64 * 2);
  _Float16* wpc116  = (_Float16*)alloc(64 * 512 * 2);
  _Float16* wpc216  = (_Float16*)alloc(32 * 64 * 2);
  _Float16* z1      = (_Float16*)alloc((size_t)BT * 512 * 2);
  _Float16* z2      = (_Float16*)alloc((size_t)BT * 256 * 2);
  _Float16* zp1     = (_Float16*)alloc((size_t)BT * 64 * 2);
  _Float16* zp2     = (_Float16*)alloc((size_t)BT * 32 * 2);
  float*    score   = (float*)alloc((size_t)BT * 4);
  float*    attn    = (float*)alloc((size_t)BT * 4);
  float*    scorep  = (float*)alloc((size_t)BT * 4);
  float*    hT      = (float*)alloc((size_t)BDIM * 256 * 4);
  float*    ipool   = (float*)alloc((size_t)BDIM * 128 * 4);
  float*    cpool   = (float*)alloc((size_t)BDIM * 128 * 4);
  float*    feat    = (float*)alloc((size_t)BDIM * 1024 * 4);
  float*    stats   = (float*)alloc(1024 * 4);

  // 1) convert weights to f16
  auto cvt = [&](const float* s, _Float16* d, int n) {
    k_cvt<<<(n + 255) / 256, 256, 0, stream>>>(s, d, n);
  };
  cvt(gru_Wih, wih16, 768 * 256);  cvt(gru_Whh, whh16, 768 * 256);
  cvt(agru_Wx, wax16, 768 * 256);  cvt(agru_Wh, wah16, 768 * 256);
  cvt(att_W1, wa116, 512 * 1024);  cvt(att_W2, wa216, 256 * 512);
  cvt(pi_W1, wpi116, 64 * 512);    cvt(pi_W2, wpi216, 32 * 64);
  cvt(pc_W1, wpc116, 64 * 512);    cvt(pc_W2, wpc216, 32 * 64);

  // 2) embeddings
  k_embed_seq<<<BT, 256, 0, stream>>>(item_seq, cate_seq, item_table, cate_table, seq16);
  k_embed_tgt<<<BDIM, 256, 0, stream>>>(tgt_item, tgt_cate, item_table, cate_table, tgt16, tgt32);

  // 3) masked GRU (interest extraction)
  k_gru<<<16, 256, 0, stream>>>(seq16, wih16, whh16, gru_bih, gru_bhh, lengths, inter16);

  // 4) DIN attention MLP with Dice, then masked softmax
  k_gemm_din<<<BT / 16, 256, 0, stream>>>(inter16, tgt16, wa116, att_b1, z1);
  k_colstats<<<512, 256, 0, stream>>>(z1, 512, stats);
  k_dice<<<BT * 512 / 256, 256, 0, stream>>>(z1, 512, stats, att_a1);
  k_gemm_plain<<<BT / 16, 256, 0, stream>>>(z1, 512, wa216, 256, att_b2, z2);
  k_colstats<<<256, 256, 0, stream>>>(z2, 256, stats);
  k_dice<<<BT * 256 / 256, 256, 0, stream>>>(z2, 256, stats, att_a2);
  k_dot<<<BT / 8, 256, 0, stream>>>(z2, 256, att_W3, att_b3, score);
  k_softmax_att<<<BDIM, 256, 0, stream>>>(score, lengths, attn);

  // 5) AGRU (interest evolution) -> hT
  k_agru<<<16, 256, 0, stream>>>(inter16, wax16, wah16, agru_bx, agru_bh, attn, lengths, hT);

  // 6) item pooling
  k_gemm_ci<<<BT / 32, 256, 0, stream>>>(seq16, tgt16, 0, wpi116, pi_b1, zp1);
  k_colstats<<<64, 256, 0, stream>>>(zp1, 64, stats);
  k_dice<<<BT * 64 / 256, 256, 0, stream>>>(zp1, 64, stats, pi_a1);
  k_gemm_plain<<<BT / 16, 256, 0, stream>>>(zp1, 64, wpi216, 32, pi_b2, zp2);
  k_colstats<<<32, 256, 0, stream>>>(zp2, 32, stats);
  k_dice<<<BT * 32 / 256, 256, 0, stream>>>(zp2, 32, stats, pi_a2);
  k_dot<<<BT / 8, 256, 0, stream>>>(zp2, 32, pi_W3, pi_b3, scorep);
  k_pool<<<BDIM, 256, 0, stream>>>(scorep, lengths, seq16, 0, ipool);

  // 7) cate pooling
  k_gemm_ci<<<BT / 32, 256, 0, stream>>>(seq16, tgt16, 128, wpc116, pc_b1, zp1);
  k_colstats<<<64, 256, 0, stream>>>(zp1, 64, stats);
  k_dice<<<BT * 64 / 256, 256, 0, stream>>>(zp1, 64, stats, pc_a1);
  k_gemm_plain<<<BT / 16, 256, 0, stream>>>(zp1, 64, wpc216, 32, pc_b2, zp2);
  k_colstats<<<32, 256, 0, stream>>>(zp2, 32, stats);
  k_dice<<<BT * 32 / 256, 256, 0, stream>>>(zp2, 32, stats, pc_a2);
  k_dot<<<BT / 8, 256, 0, stream>>>(zp2, 32, pc_W3, pc_b3, scorep);
  k_pool<<<BDIM, 256, 0, stream>>>(scorep, lengths, seq16, 128, cpool);

  // 8) final feature + MLP
  k_feat<<<BDIM * 1024 / 256, 256, 0, stream>>>(tgt32, ipool, cpool, hT, feat);
  k_mlp<<<BDIM, 256, 0, stream>>>(feat, mlp_W1, mlp_b1, mlp_W2, mlp_b2, mlp_W3, mlp_b3,
                                  (float*)d_out);
  (void)in_sizes; (void)n_in; (void)out_size; (void)ws_size;
}
